// LinearPerformerAttention_29875792511830
// MI455X (gfx1250) — matrix operand order repaired and checked by reading, not compiled, not run on hardware
//
#include <hip/hip_runtime.h>
#include <math.h>

#define DEVINL __device__ __forceinline__

typedef __attribute__((ext_vector_type(16))) __bf16 v16bf;
typedef __attribute__((ext_vector_type(8)))  __bf16 v8bf;
typedef __attribute__((ext_vector_type(8)))  float  v8f;
typedef unsigned short u16t;

// Problem constants (match reference)
static constexpr int Bc  = 4;
static constexpr int Nc  = 4096;
static constexpr int Dc  = 1024;
static constexpr int Hc  = 16;
static constexpr int Fc  = 256;
static constexpr int HDc = 64;
static constexpr int Mc  = Bc * Nc;      // 16384 token rows

// ---------------- bf16 helpers (raw-bit storage, RNE convert) ----------------
DEVINL u16t f2bf(float f) {
  unsigned u = __float_as_uint(f);
  u += 0x7fffu + ((u >> 16) & 1u);
  return (u16t)(u >> 16);
}
DEVINL float bf2f(u16t h) { return __uint_as_float(((unsigned)h) << 16); }

union Frag {
  v16bf v;
  v8bf  h[2];
  u16t  u[16];
};

DEVINL v8bf ld8(const u16t* p) { return *reinterpret_cast<const v8bf*>(p); }

// A fragment (16x32 bf16, MxK), row-major A, lane layout per ISA 7.12.2:
// lane m = L%16; lane half p = L/16; elems 0..7 <- K=p*8+j ; elems 8..15 <- K=16+p*8+j
DEVINL v16bf load_a_frag(const u16t* A, int lda) {
  const int lane = threadIdx.x & 31;
  const int m = lane & 15, p = lane >> 4;
  const u16t* r = A + (long)m * lda + p * 8;
  Frag f;
  f.h[0] = ld8(r);
  f.h[1] = ld8(r + 16);
  return f.v;
}
// A fragment where logical A(m,k) = Araw[k*lda + m] (transposed operand)
DEVINL v16bf load_a_frag_tr(const u16t* A, int lda) {
  const int lane = threadIdx.x & 31;
  const int m = lane & 15, p = lane >> 4;
  Frag f;
#pragma unroll
  for (int j = 0; j < 8; ++j) {
    f.u[j]     = A[(long)(p * 8 + j) * lda + m];
    f.u[j + 8] = A[(long)(16 + p * 8 + j) * lda + m];
  }
  return f.v;
}
// B fragment (32x16 bf16, KxN) where B is stored transposed as (N,K) row-major
// (the x @ W^T case): lane n = L%16 reads K=p*8+j and K=16+p*8+j (two 8-runs).
DEVINL v16bf load_b_frag_nk(const u16t* Bt, int ldb) {
  const int lane = threadIdx.x & 31;
  const int n = lane & 15, kg = lane >> 4;
  const u16t* r = Bt + (long)n * ldb + kg * 8;
  Frag f;
  f.h[0] = ld8(r);
  f.h[1] = ld8(r + 16);
  return f.v;
}
// B fragment where B is stored (K,N) row-major (strided per-lane gathers)
DEVINL v16bf load_b_frag_kn(const u16t* Bm, int ldb) {
  const int lane = threadIdx.x & 31;
  const int n = lane & 15, kg = lane >> 4;
  const u16t* r = Bm + (long)kg * 8 * ldb + n;
  Frag f;
#pragma unroll
  for (int j = 0; j < 16; ++j) f.u[j] = r[(long)(j < 8 ? j : j + 8) * ldb];
  return f.v;
}

DEVINL v8f wmma_bf16(v16bf a, v16bf b, v8f c) {
  // (neg_a, A, neg_b, B, c_mod, C, reuse_a, reuse_b)
  return __builtin_amdgcn_wmma_f32_16x16x32_bf16(false, a, false, b, (short)0, c,
                                                 false, false);
}

// ---------------- epilogue modes ----------------
enum Epi {
  EPI_F32 = 0,        // Cf = acc
  EPI_BF16,           // Cb = bf16(acc)
  EPI_ELU1_BF16,      // Cb = bf16(elu(acc)+1)
  EPI_GELU_BF16,      // Cb = bf16(gelu_exact(acc + bias[n]))
  EPI_BIAS_RES_F32,   // Cf = acc + bias[n] + resid[m,n]
  EPI_SCALE_BF16      // Cb = bf16(acc * rowscale[m])
};

// ---------------- tiled WMMA GEMM ----------------
// C(M,N) = A(M,K) x B(K,N). Wave tile 32x64: 2 A-frags x 4 B-frags -> 8 WMMA
// accumulators per k-step (A reused 4x, B reused 2x). Block = 4 waves stacked
// 4x1 along M -> block tile 128x64, so even N=64 GEMMs keep all waves busy.
// All dims used are exact multiples -> EXEC all-ones for every WMMA wave.
// Two-level batching: z -> (b = z/Hdiv, h = z%Hdiv) with per-level strides.
template <int EPI, bool TA, bool TB>
__global__ __launch_bounds__(128, 2) void wmma_gemm_kernel(
    const u16t* __restrict__ A, const u16t* __restrict__ Bp,
    float* __restrict__ Cf, u16t* __restrict__ Cb,
    const float* __restrict__ bias, const float* __restrict__ resid,
    const float* __restrict__ rowscale, int Mdim, int Ndim, int Kdim, int lda,
    int ldb, int ldc, int Hdiv, long sAb, long sAh, long sBb, long sBh,
    long sCb, long sCh, long sZ) {
  const int z  = blockIdx.z;
  const int bb = z / Hdiv, hh = z % Hdiv;
  A  += bb * sAb + hh * sAh;
  Bp += bb * sBb + hh * sBh;
  const long coff = bb * sCb + hh * sCh;
  if (Cf) Cf += coff;
  if (Cb) Cb += coff;
  const float* zs = rowscale ? (rowscale + (long)z * sZ) : nullptr;

  const int wave = threadIdx.x >> 5;
  const int m0 = blockIdx.y * 128 + wave * 32;
  const int n0 = blockIdx.x * 64;
  if (m0 >= Mdim || n0 >= Ndim) return;  // wave-uniform; EXEC stays all-ones

  v8f acc[2][4] = {};

  for (int k0 = 0; k0 < Kdim; k0 += 32) {
    v16bf a0, a1, bf[4];
    if (TA) {
      a0 = load_a_frag_tr(A + (long)k0 * lda + m0, lda);
      a1 = load_a_frag_tr(A + (long)k0 * lda + m0 + 16, lda);
    } else {
      a0 = load_a_frag(A + (long)m0 * lda + k0, lda);
      a1 = load_a_frag(A + (long)(m0 + 16) * lda + k0, lda);
    }
    if (TB) {
#pragma unroll
      for (int j = 0; j < 4; ++j)
        bf[j] = load_b_frag_nk(Bp + (long)(n0 + j * 16) * ldb + k0, ldb);
      if (k0 + 32 < Kdim)  // pull next weight tile toward L2 (global_prefetch_b8)
        __builtin_prefetch(Bp + (long)n0 * ldb + k0 + 32, 0, 3);
    } else {
#pragma unroll
      for (int j = 0; j < 4; ++j)
        bf[j] = load_b_frag_kn(Bp + (long)k0 * ldb + n0 + j * 16, ldb);
    }
#pragma unroll
    for (int j = 0; j < 4; ++j) {
      acc[0][j] = wmma_bf16(a0, bf[j], acc[0][j]);
      acc[1][j] = wmma_bf16(a1, bf[j], acc[1][j]);
    }
  }

  // C layout (ISA 7.12.2): lane n = L%16, VGPR r -> row (L/16)*8 + r
  const int lane  = threadIdx.x & 31;
  const int rbase = (lane >> 4) * 8;
  const int cloc  = lane & 15;
#pragma unroll
  for (int i = 0; i < 2; ++i)
#pragma unroll
    for (int j = 0; j < 4; ++j)
#pragma unroll
      for (int r = 0; r < 8; ++r) {
        const int m = m0 + i * 16 + rbase + r;
        const int n = n0 + j * 16 + cloc;
        float v = acc[i][j][r];
        const long idx = (long)m * ldc + n;
        if (EPI == EPI_F32) {
          Cf[idx] = v;
        } else if (EPI == EPI_BF16) {
          Cb[idx] = f2bf(v);
        } else if (EPI == EPI_ELU1_BF16) {
          v = v > 0.f ? v + 1.f : __expf(v);  // elu(x)+1
          Cb[idx] = f2bf(v);
        } else if (EPI == EPI_GELU_BF16) {
          v += bias[n];
          v = 0.5f * v * (1.f + erff(v * 0.70710678118654752f));
          Cb[idx] = f2bf(v);
        } else if (EPI == EPI_BIAS_RES_F32) {
          Cf[idx] = v + bias[n] + resid[idx];
        } else if (EPI == EPI_SCALE_BF16) {
          Cb[idx] = f2bf(v * zs[m]);
        }
      }
}

// ---------------- support kernels ----------------
__global__ void cvt_f32_bf16_kernel(const float* __restrict__ in,
                                    u16t* __restrict__ out, long n) {
  long i = (long)blockIdx.x * blockDim.x + threadIdx.x;
  if (i < n) out[i] = f2bf(in[i]);
}

// k_sum[z,f] = sum_n k_f[z,n,f]   (z = b*H+h, coalesced over f)
__global__ void ksum_kernel(const u16t* __restrict__ kf,
                            float* __restrict__ ksum) {
  int t = blockIdx.x * blockDim.x + threadIdx.x;  // t < 64*F
  int z = t >> 8;                                 // /F (F=256)
  int f = t & 255;
  const u16t* p = kf + (long)z * Nc * Fc + f;
  float s = 0.f;
  for (int n = 0; n < Nc; ++n) s += bf2f(p[(long)n * Fc]);
  ksum[t] = s;
}

// z[z,n] = 1 / (q_f[z,n,:] . k_sum[z,:] + eps); one wave32 per row
__global__ __launch_bounds__(256) void zden_kernel(const u16t* __restrict__ qf,
                                                   const float* __restrict__ ksum,
                                                   float* __restrict__ zout) {
  int row  = blockIdx.x * 8 + (threadIdx.x >> 5);  // [0, 64*N)
  int lane = threadIdx.x & 31;
  int zb   = row >> 12;  // /N
  const u16t*  p  = qf + (long)row * Fc;
  const float* ks = ksum + (long)zb * Fc;
  float s = 0.f;
  for (int f = lane; f < Fc; f += 32) s += bf2f(p[f]) * ks[f];
#pragma unroll
  for (int off = 16; off > 0; off >>= 1) s += __shfl_xor(s, off, 32);
  if (lane == 0) zout[row] = 1.0f / (s + 1e-8f);
}

// LayerNorm over D=1024; one 256-thread block per row; optional bf16 mirror
__global__ __launch_bounds__(256) void layernorm_kernel(
    const float* __restrict__ in, const float* __restrict__ gamma,
    const float* __restrict__ beta, float* __restrict__ outF,
    u16t* __restrict__ outB) {
  __shared__ float shs[8], shs2[8];
  const long row = blockIdx.x;
  const float* p = in + row * Dc;
  float v[4], s = 0.f, s2 = 0.f;
#pragma unroll
  for (int i = 0; i < 4; ++i) {
    v[i] = p[threadIdx.x + i * 256];
    s += v[i];
    s2 += v[i] * v[i];
  }
  const int wave = threadIdx.x >> 5, lane = threadIdx.x & 31;
#pragma unroll
  for (int off = 16; off > 0; off >>= 1) {
    s  += __shfl_xor(s, off, 32);
    s2 += __shfl_xor(s2, off, 32);
  }
  if (lane == 0) { shs[wave] = s; shs2[wave] = s2; }
  __syncthreads();
  if (threadIdx.x == 0) {
    float a = 0.f, b = 0.f;
#pragma unroll
    for (int w = 0; w < 8; ++w) { a += shs[w]; b += shs2[w]; }
    shs[0] = a; shs2[0] = b;
  }
  __syncthreads();
  const float mu  = shs[0] * (1.f / Dc);
  const float var = shs2[0] * (1.f / Dc) - mu * mu;
  const float rs  = rsqrtf(var + 1e-5f);
#pragma unroll
  for (int i = 0; i < 4; ++i) {
    const int c = threadIdx.x + i * 256;
    const float o = (v[i] - mu) * rs * gamma[c] + beta[c];
    outF[row * Dc + c] = o;
    if (outB) outB[row * Dc + c] = f2bf(o);
  }
}

// ---------------- host orchestration ----------------
extern "C" void kernel_launch(void* const* d_in, const int* in_sizes, int n_in,
                              void* d_out, int out_size, void* d_ws,
                              size_t ws_size, hipStream_t stream) {
  (void)in_sizes; (void)n_in; (void)out_size; (void)ws_size;
  const float* x     = (const float*)d_in[0];
  const float* proj  = (const float*)d_in[1];
  const float* w_qkv = (const float*)d_in[2];
  const float* w_out = (const float*)d_in[3];
  const float* b_out = (const float*)d_in[4];
  const float* gamma = (const float*)d_in[5];
  const float* beta  = (const float*)d_in[6];
  const float* w1    = (const float*)d_in[7];
  const float* b1    = (const float*)d_in[8];
  const float* w2    = (const float*)d_in[9];
  const float* b2    = (const float*)d_in[10];
  float* out = (float*)d_out;

  // Aliased workspace arena (lifetimes are strictly sequential on `stream`)
  char* ws = (char*)d_ws;
  const size_t MBy = 1ull << 20;
  u16t*  qfb   = (u16t*)(ws + 0);           // 128MB: q_f bf16
  u16t*  hb    = qfb;                       //   -> reused: FFN hidden (128MB)
  u16t*  kfb   = (u16t*)(ws + 128 * MBy);   // 128MB: k_f bf16
  float* y1    = (float*)(ws + 128 * MBy);  //   -> reused: pre-LN1 f32 (64MB)
  float* y1n   = (float*)(ws + 192 * MBy);  //   -> reused: post-LN1 f32 (64MB)
  float* y2    = y1;                        //   -> reused: pre-LN2 f32 (64MB)
  u16t*  qkvb  = (u16t*)(ws + 256 * MBy);   // 96MB: qkv bf16
  u16t*  attnm = qkvb;                      //   -> reused: merged attn bf16 (32MB)
  u16t*  x1b   = (u16t*)(ws + 288 * MBy);   //   -> reused: post-LN1 bf16 (32MB)
  u16t*  xb    = (u16t*)(ws + 352 * MBy);   // 32MB: x bf16
  u16t*  kvb   = (u16t*)(ws + 352 * MBy);   //   -> reused: kv bf16 (2MB)
  float* zbuf  = (float*)(ws + 356 * MBy);  //   -> reused: z (1MB)
  float* ksum  = (float*)(ws + 360 * MBy);  //   -> reused: k_sum (64KB)
  u16t*  wqkvb = (u16t*)(ws + 384 * MBy);   // 6MB
  u16t*  projb = (u16t*)(ws + 390 * MBy);   // 0.5MB
  u16t*  woutb = (u16t*)(ws + 391 * MBy);   // 2MB
  u16t*  w1b   = (u16t*)(ws + 393 * MBy);   // 8MB
  u16t*  w2b   = (u16t*)(ws + 401 * MBy);   // 8MB -> total ~409MB

  auto cdiv = [](long a, long b) { return (int)((a + b - 1) / b); };
  const dim3 gblk(128);

  // ---- 0) f32 -> bf16 conversions
  cvt_f32_bf16_kernel<<<cdiv((long)Mc * Dc, 256), 256, 0, stream>>>(x, xb, (long)Mc * Dc);
  cvt_f32_bf16_kernel<<<cdiv(3L * Dc * Dc, 256), 256, 0, stream>>>(w_qkv, wqkvb, 3L * Dc * Dc);
  cvt_f32_bf16_kernel<<<cdiv((long)Hc * HDc * Fc, 256), 256, 0, stream>>>(proj, projb, (long)Hc * HDc * Fc);
  cvt_f32_bf16_kernel<<<cdiv((long)Dc * Dc, 256), 256, 0, stream>>>(w_out, woutb, (long)Dc * Dc);
  cvt_f32_bf16_kernel<<<cdiv(4L * Dc * Dc, 256), 256, 0, stream>>>(w1, w1b, 4L * Dc * Dc);
  cvt_f32_bf16_kernel<<<cdiv(4L * Dc * Dc, 256), 256, 0, stream>>>(w2, w2b, 4L * Dc * Dc);

  // ---- 1) qkv = x @ w_qkv^T            (16384 x 3072 x 1024) -> bf16
  wmma_gemm_kernel<EPI_BF16, false, true>
      <<<dim3(3 * Dc / 64, Mc / 128, 1), gblk, 0, stream>>>(
          xb, wqkvb, nullptr, qkvb, nullptr, nullptr, nullptr,
          Mc, 3 * Dc, Dc, Dc, Dc, 3 * Dc, 1, 0, 0, 0, 0, 0, 0, 0);

  // ---- 2) q_f = elu(q @ proj_h)+1      batched over (b,h): 4096 x 256 x 64
  wmma_gemm_kernel<EPI_ELU1_BF16, false, false>
      <<<dim3(Fc / 64, Nc / 128, Bc * Hc), gblk, 0, stream>>>(
          qkvb /*q cols*/, projb, nullptr, qfb, nullptr, nullptr, nullptr,
          Nc, Fc, HDc, 3 * Dc, Fc, Fc, Hc,
          (long)Nc * 3 * Dc, (long)HDc,            // A: b-row blocks, head cols
          0, (long)HDc * Fc,                       // B: proj per head
          (long)Hc * Nc * Fc, (long)Nc * Fc, 0);   // C: z-contiguous

  // ---- 3) k_f = elu(k @ proj_h)+1
  wmma_gemm_kernel<EPI_ELU1_BF16, false, false>
      <<<dim3(Fc / 64, Nc / 128, Bc * Hc), gblk, 0, stream>>>(
          qkvb + Dc /*k cols*/, projb, nullptr, kfb, nullptr, nullptr, nullptr,
          Nc, Fc, HDc, 3 * Dc, Fc, Fc, Hc,
          (long)Nc * 3 * Dc, (long)HDc, 0, (long)HDc * Fc,
          (long)Hc * Nc * Fc, (long)Nc * Fc, 0);

  // ---- 4) k_sum and z denominators
  ksum_kernel<<<Bc * Hc * Fc / 256, 256, 0, stream>>>(kfb, ksum);
  zden_kernel<<<Bc * Hc * Nc / 8, 256, 0, stream>>>(qfb, ksum, zbuf);

  // ---- 5) kv = k_f^T @ v               batched: 256 x 64 x 4096 -> bf16
  wmma_gemm_kernel<EPI_BF16, true, false>
      <<<dim3(HDc / 64, Fc / 128, Bc * Hc), gblk, 0, stream>>>(
          kfb, qkvb + 2 * Dc /*v cols*/, nullptr, kvb, nullptr, nullptr, nullptr,
          Fc, HDc, Nc, Fc /*lda of raw k_f*/, 3 * Dc, HDc, Hc,
          (long)Hc * Nc * Fc, (long)Nc * Fc,
          (long)Nc * 3 * Dc, (long)HDc,
          (long)Hc * Fc * HDc, (long)Fc * HDc, 0);

  // ---- 6) attn = (q_f @ kv) * z        batched: 4096 x 64 x 256, scatter to (B,N,D)
  wmma_gemm_kernel<EPI_SCALE_BF16, false, false>
      <<<dim3(1, Nc / 128, Bc * Hc), gblk, 0, stream>>>(
          qfb, kvb, nullptr, attnm, nullptr, nullptr, zbuf,
          Nc, HDc, Fc, Fc, HDc, Dc, Hc,
          (long)Hc * Nc * Fc, (long)Nc * Fc,
          (long)Hc * Fc * HDc, (long)Fc * HDc,
          (long)Nc * Dc, (long)HDc, (long)Nc);

  // ---- 7) y1 = x + attn @ w_out^T + b_out   (16384 x 1024 x 1024) -> f32
  wmma_gemm_kernel<EPI_BIAS_RES_F32, false, true>
      <<<dim3(Dc / 64, Mc / 128, 1), gblk, 0, stream>>>(
          attnm, woutb, y1, nullptr, b_out, x, nullptr,
          Mc, Dc, Dc, Dc, Dc, Dc, 1, 0, 0, 0, 0, 0, 0, 0);

  // ---- 8) LN1 -> y1n (f32) + x1b (bf16)
  layernorm_kernel<<<Mc, 256, 0, stream>>>(y1, gamma, beta, y1n, x1b);

  // ---- 9) h = gelu(x1 @ w1^T + b1)     (16384 x 4096 x 1024) -> bf16
  wmma_gemm_kernel<EPI_GELU_BF16, false, true>
      <<<dim3(4 * Dc / 64, Mc / 128, 1), gblk, 0, stream>>>(
          x1b, w1b, nullptr, hb, b1, nullptr, nullptr,
          Mc, 4 * Dc, Dc, Dc, Dc, 4 * Dc, 1, 0, 0, 0, 0, 0, 0, 0);

  // ---- 10) y2 = y1n + h @ w2^T + b2    (16384 x 1024 x 4096) -> f32
  wmma_gemm_kernel<EPI_BIAS_RES_F32, false, true>
      <<<dim3(Dc / 64, Mc / 128, 1), gblk, 0, stream>>>(
          hb, w2b, y2, nullptr, b2, y1n, nullptr,
          Mc, Dc, 4 * Dc, 4 * Dc, 4 * Dc, Dc, 1, 0, 0, 0, 0, 0, 0, 0);

  // ---- 11) LN2 -> d_out
  layernorm_kernel<<<Mc, 256, 0, stream>>>(y2, gamma, beta, out, nullptr);
}
